// GraphSAGE_5085241279053
// MI455X (gfx1250) — matrix operand order repaired and checked by reading, not compiled, not run on hardware
//
#include <hip/hip_runtime.h>
#include <hip/hip_bf16.h>
#include <stdint.h>

typedef __attribute__((ext_vector_type(16))) __bf16 v16bf;
typedef __attribute__((ext_vector_type(8)))  float  v8f;

#define N_NODES 100000
#define N_EDGES 3200000
#define F_IN    1433
#define KPAD2   1536      // F_IN padded so KC divides it (power-of-2 staging math)
#define KEFF    1440      // first multiple of 32 >= F_IN; WMMA loop stops here
#define HID     32
#define KC      256       // K panel staged in LDS per chunk
#define LSTRIDE 264       // LDS panel row stride in bf16 (528 B = 132 dw; 132%64=4 -> conflict-free)
#define LDS_BYTES (64 * LSTRIDE * 2)   // 33792

// ---- f32 -> bf16, round-to-nearest-even ------------------------------------
__device__ __forceinline__ __bf16 f2bf(float f) {
  union { float f; uint32_t u; } c; c.f = f;
  uint32_t r = c.u + 0x7FFFu + ((c.u >> 16) & 1u);
  uint16_t h = (uint16_t)(r >> 16);
  return __builtin_bit_cast(__bf16, h);
}

// ---- pack [Wl|Wr] (f32) into bf16 panels: Wb0[64][KPAD2], Wb1[64][HID] -----
__global__ void prep_weights(const float* __restrict__ Wl0, const float* __restrict__ Wr0,
                             const float* __restrict__ Wl1, const float* __restrict__ Wr1,
                             __bf16* __restrict__ Wb0, __bf16* __restrict__ Wb1) {
  int gid = blockIdx.x * blockDim.x + threadIdx.x;
  const int total0 = 64 * KPAD2;
  if (gid < total0) {
    int n = gid / KPAD2;
    int k = gid - n * KPAD2;
    float v = 0.f;
    if (k < F_IN) v = (n < HID) ? Wl0[n * F_IN + k] : Wr0[(n - HID) * F_IN + k];
    Wb0[gid] = f2bf(v);
  } else {
    int g = gid - total0;
    if (g < 64 * HID) {
      int n = g >> 5, k = g & 31;
      float v = (n < HID) ? Wl1[n * HID + k] : Wr1[(n - HID) * HID + k];
      Wb1[g] = f2bf(v);
    }
  }
}

union BU { uint4 q[2]; v16bf v; };

// ---- Layer-1 dual GEMM: hlin = X@Wl.T, self = X@Wr.T  (bf16 WMMA, f32 acc) -
// 256 threads = 8 waves; each wave: 16 rows x 64 cols; K loop 1440 in 32-chunks.
__global__ __launch_bounds__(256) void gemm1_wmma(
    const float* __restrict__ X, const __bf16* __restrict__ Wb0,
    float* __restrict__ hlin, float* __restrict__ selfo) {
  extern __shared__ uint4 ldsu4[];          // 64 rows x LSTRIDE bf16 (padded)
  const char* lwB = (const char*)ldsu4;

  const int tid   = threadIdx.x;
  const int lane  = tid & 31;
  const int wave  = tid >> 5;
  const int lanlo = lane & 15;
  const int hf    = lane >> 4;

  const int rowBase = blockIdx.x * 128 + wave * 16;
  int rowA = rowBase + lanlo;
  if (rowA >= N_NODES) rowA = N_NODES - 1;  // clamp reads; stores predicated
  const float* xrow = X + (size_t)rowA * F_IN;

  // per-tile LDS byte base for B fragments (col = t*16+lanlo, K slot = hf*16)
  int bbase[4];
#pragma unroll
  for (int t = 0; t < 4; ++t)
    bbase[t] = ((t * 16 + lanlo) * LSTRIDE + hf * 16) * 2;

  v8f acc0 = {}, acc1 = {}, acc2 = {}, acc3 = {};

  for (int kc0 = 0; kc0 < KEFF; kc0 += KC) {
    __syncthreads();
    // stage 64 x KC bf16 panel into LDS with padded row stride (shift/mask idx)
    const uint4* gsrc = (const uint4*)Wb0;   // row stride KPAD2/8 = 192 uint4
#pragma unroll
    for (int j = 0; j < 8; ++j) {
      int i = tid + j * 256;                 // 0..2047
      int n = i >> 5, q = i & 31;            // KC/8 = 32 uint4 per row
      ldsu4[n * 33 + q] = gsrc[n * (KPAD2 / 8) + (kc0 >> 3) + q];  // 33 uint4 = LSTRIDE*2/16
    }
    __syncthreads();

    const int kmax = (KEFF - kc0 < KC) ? (KEFF - kc0) : KC;  // 256, or 160 last

    // prologue: B fragments for kk = 0
    BU b0, b1, b2, b3;
    b0.q[0] = *(const uint4*)(lwB + bbase[0]);
    b0.q[1] = *(const uint4*)(lwB + bbase[0] + 16);
    b1.q[0] = *(const uint4*)(lwB + bbase[1]);
    b1.q[1] = *(const uint4*)(lwB + bbase[1] + 16);
    b2.q[0] = *(const uint4*)(lwB + bbase[2]);
    b2.q[1] = *(const uint4*)(lwB + bbase[2] + 16);
    b3.q[0] = *(const uint4*)(lwB + bbase[3]);
    b3.q[1] = *(const uint4*)(lwB + bbase[3] + 16);

#pragma unroll 4
    for (int kk = 0; kk < kmax; kk += 32) {
      const int k0 = kc0 + kk;
      // --- A fragment (16x32 bf16), contiguous layout: lane half h K={16h..16h+15}+k0
      v16bf a;
      const int kb = k0 + hf * 16;
      if (k0 + 32 <= F_IN) {        // uniform fast path (44 of 45 steps)
#pragma unroll
        for (int i = 0; i < 8; ++i) a[i]     = f2bf(xrow[kb + i]);
#pragma unroll
        for (int i = 0; i < 8; ++i) a[8 + i] = f2bf(xrow[kb + 8 + i]);
      } else {                      // tail: clamp address, select 0 (no exec branches)
#pragma unroll
        for (int i = 0; i < 8; ++i) {
          int k = kb + i;
          float v = xrow[k < F_IN ? k : 0];
          a[i] = f2bf(k < F_IN ? v : 0.f);
        }
#pragma unroll
        for (int i = 0; i < 8; ++i) {
          int k = kb + 8 + i;
          float v = xrow[k < F_IN ? k : 0];
          a[8 + i] = f2bf(k < F_IN ? v : 0.f);
        }
      }

      // prefetch next step's B fragments (clamped; harmless dup on last step)
      const int kn = (kk + 32 < kmax) ? (kk + 32) * 2 : kk * 2;
      BU n0, n1, n2, n3;
      n0.q[0] = *(const uint4*)(lwB + bbase[0] + kn);
      n0.q[1] = *(const uint4*)(lwB + bbase[0] + kn + 16);
      n1.q[0] = *(const uint4*)(lwB + bbase[1] + kn);
      n1.q[1] = *(const uint4*)(lwB + bbase[1] + kn + 16);
      n2.q[0] = *(const uint4*)(lwB + bbase[2] + kn);
      n2.q[1] = *(const uint4*)(lwB + bbase[2] + kn + 16);
      n3.q[0] = *(const uint4*)(lwB + bbase[3] + kn);
      n3.q[1] = *(const uint4*)(lwB + bbase[3] + kn + 16);

      acc0 = __builtin_amdgcn_wmma_f32_16x16x32_bf16(false, a, false, b0.v, (short)0, acc0, false, false);
      acc1 = __builtin_amdgcn_wmma_f32_16x16x32_bf16(false, a, false, b1.v, (short)0, acc1, false, false);
      acc2 = __builtin_amdgcn_wmma_f32_16x16x32_bf16(false, a, false, b2.v, (short)0, acc2, false, false);
      acc3 = __builtin_amdgcn_wmma_f32_16x16x32_bf16(false, a, false, b3.v, (short)0, acc3, false, false);

      b0 = n0; b1 = n1; b2 = n2; b3 = n3;
    }
  }

  // store: tiles 0,1 -> hlin (cols 0..31); tiles 2,3 -> self (cols 0..31)
#pragma unroll
  for (int t = 0; t < 4; ++t) {
    const v8f acc = (t == 0) ? acc0 : (t == 1) ? acc1 : (t == 2) ? acc2 : acc3;
    const int col = ((t & 1) * 16) + lanlo;
    float* outp = (t < 2) ? hlin : selfo;
#pragma unroll
    for (int r = 0; r < 8; ++r) {
      int row = rowBase + hf * 8 + r;
      if (row < N_NODES) outp[(size_t)row * HID + col] = acc[r];
    }
  }
}

// ---- Layer-2 dual GEMM (K=32): one WMMA per col tile ------------------------
__global__ __launch_bounds__(256) void gemm2_wmma(
    const float* __restrict__ H, const __bf16* __restrict__ Wb1,
    float* __restrict__ hlin, float* __restrict__ selfo) {
  const int tid = threadIdx.x, lane = tid & 31, wave = tid >> 5;
  const int lanlo = lane & 15, hf = lane >> 4;
  const int rowBase = blockIdx.x * 128 + wave * 16;
  int rowA = rowBase + lanlo;
  if (rowA >= N_NODES) rowA = N_NODES - 1;

  // rows are 128B aligned -> float4 loads; fragment K slots {kb..kb+15}
  const float4* hrow = (const float4*)(H + (size_t)rowA * HID);
  const int kb4 = hf * 4;
  float4 s0 = hrow[kb4], s1 = hrow[kb4 + 1], s2 = hrow[kb4 + 2], s3 = hrow[kb4 + 3];
  v16bf a;
  a[0]  = f2bf(s0.x); a[1]  = f2bf(s0.y); a[2]  = f2bf(s0.z); a[3]  = f2bf(s0.w);
  a[4]  = f2bf(s1.x); a[5]  = f2bf(s1.y); a[6]  = f2bf(s1.z); a[7]  = f2bf(s1.w);
  a[8]  = f2bf(s2.x); a[9]  = f2bf(s2.y); a[10] = f2bf(s2.z); a[11] = f2bf(s2.w);
  a[12] = f2bf(s3.x); a[13] = f2bf(s3.y); a[14] = f2bf(s3.z); a[15] = f2bf(s3.w);

#pragma unroll
  for (int t = 0; t < 4; ++t) {
    const v16bf b = *(const v16bf*)(Wb1 + (t * 16 + lanlo) * HID + hf * 16);
    v8f acc = {};
    acc = __builtin_amdgcn_wmma_f32_16x16x32_bf16(false, a, false, b,
                                                  (short)0, acc, false, false);
    const int col = ((t & 1) * 16) + lanlo;
    float* outp = (t < 2) ? hlin : selfo;
#pragma unroll
    for (int r = 0; r < 8; ++r) {
      int row = rowBase + hf * 8 + r;
      if (row < N_NODES) outp[(size_t)row * HID + col] = acc[r];
    }
  }
}

// ---- edge scatter: all 32 lanes of a wave share one edge (scalar src/dst) --
__global__ void scatter_feat(const float* __restrict__ h, const int* __restrict__ src,
                             const int* __restrict__ dst, float* __restrict__ agg) {
  int gid = blockIdx.x * blockDim.x + threadIdx.x;
  if (gid >= N_EDGES * 32) return;
  int e = gid >> 5, c = gid & 31;
  atomicAdd(agg + (size_t)dst[e] * HID + c, h[(size_t)src[e] * HID + c]);
}

__global__ void count_edges(const int* __restrict__ dst, float* __restrict__ cnt) {
  int e = blockIdx.x * blockDim.x + threadIdx.x;
  if (e < N_EDGES) atomicAdd(cnt + dst[e], 1.0f);
}

// ---- combine layer 1: mean + self + bias ------------------------------------
__global__ void combine1(const float* __restrict__ agg, const float* __restrict__ cnt,
                         const float* __restrict__ selfo, const float* __restrict__ b,
                         float* __restrict__ h1) {
  int gid = blockIdx.x * blockDim.x + threadIdx.x;
  if (gid >= N_NODES * HID) return;
  int i = gid >> 5, c = gid & 31;
  float cn = cnt[i];
  float mean = (cn > 0.f) ? agg[gid] / cn : 0.f;
  h1[gid] = mean + selfo[gid] + b[c];
}

// ---- combine layer 2 + ReLU + log_softmax (wave32: lane == column) ----------
__global__ __launch_bounds__(256) void combine2_softmax(
    const float* __restrict__ agg, const float* __restrict__ cnt,
    const float* __restrict__ selfo, const float* __restrict__ b,
    float* __restrict__ out) {
  int lane = threadIdx.x & 31, wave = threadIdx.x >> 5;
  int row = blockIdx.x * 8 + wave;
  if (row >= N_NODES) return;
  int idx = row * HID + lane;
  float cn = cnt[row];
  float mean = (cn > 0.f) ? agg[idx] / cn : 0.f;
  float v = mean + selfo[idx] + b[lane];
  v = fmaxf(v, 0.f);
  float m = v;
  for (int off = 16; off > 0; off >>= 1) m = fmaxf(m, __shfl_xor(m, off, 32));
  float s = __expf(v - m);
  for (int off = 16; off > 0; off >>= 1) s += __shfl_xor(s, off, 32);
  out[idx] = (v - m) - __logf(s);
}

extern "C" void kernel_launch(void* const* d_in, const int* in_sizes, int n_in,
                              void* d_out, int out_size, void* d_ws, size_t ws_size,
                              hipStream_t stream) {
  (void)in_sizes; (void)n_in; (void)out_size; (void)ws_size;
  const float* x   = (const float*)d_in[0];
  const int*   ei  = (const int*)d_in[1];
  const float* Wl0 = (const float*)d_in[2];
  const float* Wr0 = (const float*)d_in[3];
  const float* b0  = (const float*)d_in[4];
  const float* Wl1 = (const float*)d_in[5];
  const float* Wr1 = (const float*)d_in[6];
  const float* b1  = (const float*)d_in[7];
  const int* src = ei;
  const int* dst = ei + N_EDGES;
  float* out = (float*)d_out;

  // workspace carve-up (256B aligned)
  char* ws = (char*)d_ws;
  size_t off = 0;
  auto take = [&](size_t bytes) -> char* {
    char* p = ws + off;
    off = (off + bytes + 255) & ~(size_t)255;
    return p;
  };
  const size_t NF = (size_t)N_NODES * HID * sizeof(float);
  __bf16* Wb0  = (__bf16*)take((size_t)64 * KPAD2 * 2);
  __bf16* Wb1  = (__bf16*)take((size_t)64 * HID * 2);
  float* hlin0 = (float*)take(NF);
  float* self0 = (float*)take(NF);
  float* agg1  = (float*)take(NF);
  float* cnt   = (float*)take((size_t)N_NODES * sizeof(float));
  float* h1    = (float*)take(NF);
  float* hlin1 = (float*)take(NF);
  float* self1 = (float*)take(NF);
  float* agg2  = (float*)take(NF);

  (void)hipMemsetAsync(agg1, 0, NF, stream);
  (void)hipMemsetAsync(agg2, 0, NF, stream);
  (void)hipMemsetAsync(cnt, 0, (size_t)N_NODES * sizeof(float), stream);

  prep_weights<<<(64 * KPAD2 + 64 * HID + 255) / 256, 256, 0, stream>>>(
      Wl0, Wr0, Wl1, Wr1, Wb0, Wb1);
  count_edges<<<(N_EDGES + 255) / 256, 256, 0, stream>>>(dst, cnt);

  const int nbRows = (N_NODES + 127) / 128;
  gemm1_wmma<<<nbRows, 256, LDS_BYTES, stream>>>(x, Wb0, hlin0, self0);
  scatter_feat<<<(N_EDGES * 32 + 255) / 256, 256, 0, stream>>>(hlin0, src, dst, agg1);
  combine1<<<(N_NODES * HID + 255) / 256, 256, 0, stream>>>(agg1, cnt, self0, b0, h1);

  gemm2_wmma<<<nbRows, 256, 0, stream>>>(h1, Wb1, hlin1, self1);
  scatter_feat<<<(N_EDGES * 32 + 255) / 256, 256, 0, stream>>>(hlin1, src, dst, agg2);
  combine2_softmax<<<(N_NODES + 7) / 8, 256, 0, stream>>>(agg2, cnt, self1, b1, out);
}
